// FlashMHA_80539226734715
// MI455X (gfx1250) — matrix operand order repaired and checked by reading, not compiled, not run on hardware
//
#include <hip/hip_runtime.h>
#include <hip/hip_bf16.h>

// ---------------------------------------------------------------------------
// FlashMHA forward for MI455X (gfx1250, wave32, WMMA + Tensor Data Mover)
//   prep : query -> bf16 ; W_qkv, W_out -> bf16 transposed [N][K]
//   qkv  = q_bf16 @ W_qkv + b_qkv      (bf16 WMMA, f32 accum, TDM-fed LDS)
//   attn = softmax(q k^T / 8) v        (flash, bf16 WMMA, TDM Q/K tiles)
//   out  = attn @ W_out + b_out        (bf16 WMMA, f32 accum, f32 out)
// ---------------------------------------------------------------------------

#define EMBED_DIM 1024
#define NUM_HEADS 16
#define HEAD_DIM  64
#define SEQ       2048
#define BATCH     2

typedef __attribute__((ext_vector_type(16))) __bf16 v16bf;
typedef __attribute__((ext_vector_type(8)))  float  v8f;
typedef __attribute__((ext_vector_type(4)))  unsigned uint32x4;
typedef __attribute__((ext_vector_type(8)))  int      int32x8;
typedef __attribute__((ext_vector_type(4)))  int      int32x4;

union FragBF { v16bf v; unsigned u[8]; };

// --- Tensor Data Mover: 2D tile (global, element-strided) -> LDS (packed) ---
// D# built per CDNA5 ISA 8.3/8.4. tensor_dim == tile_dim (tiles never OOB).
// data_size = 2 bytes (bf16). Groups 2/3 zero (2D tensor, no iterate/gather).
// This toolchain's builtin arity is 6 (clang-23 / therock-10.0 headers).
__device__ inline void tdm_load_2d_bf16(unsigned lds_off, const __bf16* gptr,
                                        unsigned tile0, unsigned tile1,
                                        unsigned long long stride0_elems) {
  const unsigned long long ga = (unsigned long long)gptr;
  uint32x4 g0;
  g0[0] = 1u;                                   // count=1, user descriptor
  g0[1] = lds_off;                              // LDS byte address
  g0[2] = (unsigned)(ga & 0xFFFFFFFFu);         // global_addr[31:0]
  g0[3] = (unsigned)((ga >> 32) & 0x1FFFFFFu) | (2u << 30);  // [56:32] | type=2
  int32x8 g1;
  g1[0] = (int)(1u << 16);                      // data_size=1 -> 2 bytes
  g1[1] = (int)((tile0 & 0xFFFFu) << 16);       // tensor_dim0[15:0] (==tile0)
  g1[2] = (int)((tile0 >> 16) & 0xFFFFu) | (int)((tile1 & 0xFFFFu) << 16);
  g1[3] = (int)((tile1 >> 16) & 0xFFFFu) | (int)((tile0 & 0xFFFFu) << 16); // tile_dim0
  g1[4] = (int)(tile1 & 0xFFFFu);               // tile_dim1 ; tile_dim2=0
  g1[5] = (int)(stride0_elems & 0xFFFFFFFFull); // tensor_dim0_stride[31:0]
  g1[6] = (int)((stride0_elems >> 32) & 0xFFFFull); // [47:32] ; dim1_stride=0
  g1[7] = 0;
  const int32x4 z4 = {0, 0, 0, 0};
  const int32x8 z8 = {0, 0, 0, 0, 0, 0, 0, 0};
  __builtin_amdgcn_tensor_load_to_lds(g0, g1, z4, z4, z8, 0);
}

__device__ inline unsigned lds_offset_of(const void* p) {
  return (unsigned)(unsigned long long)p;       // addr[31:0] == LDS offset
}

// --- WMMA operand loaders (CDNA5 16-bit layouts, wave32) --------------------
__device__ inline v16bf load_frag_a(const __bf16* smem, int row0, int k0,
                                    int ld, int lane) {
  const int m  = lane & 15;
  const int kh = (lane >> 4) << 3;
  FragBF f;
  const unsigned* p0 = (const unsigned*)(smem + (row0 + m) * ld + k0 + kh);
  const unsigned* p1 = (const unsigned*)(smem + (row0 + m) * ld + k0 + 16 + kh);
#pragma unroll
  for (int j = 0; j < 4; ++j) { f.u[j] = p0[j]; f.u[4 + j] = p1[j]; }
  return f.v;
}

__device__ inline v16bf load_frag_b(const __bf16* smem, int n0, int k0,
                                    int ld, int lane) {
  const int n  = lane & 15;
  const int kh = (lane >> 4) << 3;
  FragBF f;
  const unsigned* p0 = (const unsigned*)(smem + (n0 + n) * ld + k0 + kh);
  const unsigned* p1 = (const unsigned*)(smem + (n0 + n) * ld + k0 + 16 + kh);
#pragma unroll
  for (int j = 0; j < 4; ++j) { f.u[j] = p0[j]; f.u[4 + j] = p1[j]; }
  return f.v;
}

#define WMMA_BF16(A, B, C) \
  __builtin_amdgcn_wmma_f32_16x16x32_bf16(false, (A), false, (B), (short)0, (C), false, false)

// ---------------------------------------------------------------------------
// Prep kernels (one-shot, bandwidth trivial vs the GEMMs)
// ---------------------------------------------------------------------------
__global__ __launch_bounds__(256)
void cvt_f32_to_bf16(const float* __restrict__ in, __bf16* __restrict__ out) {
  const size_t i = ((size_t)blockIdx.x * 256 + threadIdx.x) * 4;
  float4 v = *(const float4*)(in + i);
  out[i + 0] = (__bf16)v.x; out[i + 1] = (__bf16)v.y;
  out[i + 2] = (__bf16)v.z; out[i + 3] = (__bf16)v.w;
}

// in: [R][C] f32  ->  out: [C][R] bf16  (weights pre-transposed for TDM B-feed)
__global__ __launch_bounds__(256)
void cvt_transpose_bf16(const float* __restrict__ in, __bf16* __restrict__ out,
                        int R, int C) {
  __shared__ float tile[32][33];
  const int r0 = blockIdx.y * 32, c0 = blockIdx.x * 32;
  const int tx = threadIdx.x & 31, ty = threadIdx.x >> 5;   // 32 x 8
#pragma unroll
  for (int i = 0; i < 32; i += 8)
    tile[ty + i][tx] = in[(size_t)(r0 + ty + i) * C + c0 + tx];
  __syncthreads();
#pragma unroll
  for (int i = 0; i < 32; i += 8)
    out[(size_t)(c0 + ty + i) * R + r0 + tx] = (__bf16)tile[tx][ty + i];
}

// ---------------------------------------------------------------------------
// GEMM: C[M,N] = A[M,K] @ B[K,N] + bias[N]
//   A bf16 [M][K] row-major; BT bf16 [N][K] (pre-transposed weights)
//   Block 256 thr (8 waves), tile 128x64x64, TDM double-buffered LDS.
// ---------------------------------------------------------------------------
#define BM 128
#define BN 64
#define BK 64
#define GEMM_THREADS 256

template <bool OUT_BF16>
__global__ __launch_bounds__(GEMM_THREADS)
void mha_gemm_tdm_wmma(const __bf16* __restrict__ A, const __bf16* __restrict__ BT,
                       const float* __restrict__ bias, void* __restrict__ Cout,
                       int M, int N, int K) {
  __shared__ __bf16 As[2][BM][BK];   // 32 KB, TDM-filled
  __shared__ __bf16 Bs[2][BN][BK];   // 16 KB, TDM-filled ([n][k])

  const int tid  = threadIdx.x;
  const int lane = tid & 31;
  const int wave = tid >> 5;
  const int wm   = wave & 3;
  const int wn   = wave >> 2;
  const int tileM = blockIdx.y * BM;
  const int tileN = blockIdx.x * BN;
  const int nk = K / BK;

  v8f acc[2][2];
#pragma unroll
  for (int i = 0; i < 2; ++i)
#pragma unroll
    for (int j = 0; j < 2; ++j) acc[i][j] = (v8f){0.f,0.f,0.f,0.f,0.f,0.f,0.f,0.f};

  // Preload k-tile 0 (wave 0 drives the TDM; barrier hands off to all waves)
  if (wave == 0) {
    tdm_load_2d_bf16(lds_offset_of(&As[0][0][0]),
                     A + (size_t)tileM * K, BK, BM, (unsigned long long)K);
    tdm_load_2d_bf16(lds_offset_of(&Bs[0][0][0]),
                     BT + (size_t)tileN * K, BK, BN, (unsigned long long)K);
    __builtin_amdgcn_s_wait_tensorcnt(0);
  }
  __syncthreads();

  for (int kt = 0; kt < nk; ++kt) {
    const int cur = kt & 1, nxt = cur ^ 1;
    if (wave == 0 && kt + 1 < nk) {          // async-prefetch next tile
      tdm_load_2d_bf16(lds_offset_of(&As[nxt][0][0]),
                       A + (size_t)tileM * K + (kt + 1) * BK, BK, BM,
                       (unsigned long long)K);
      tdm_load_2d_bf16(lds_offset_of(&Bs[nxt][0][0]),
                       BT + (size_t)tileN * K + (kt + 1) * BK, BK, BN,
                       (unsigned long long)K);
    }
#pragma unroll
    for (int ks = 0; ks < 2; ++ks) {
      v16bf a0 = load_frag_a(&As[cur][0][0], wm * 32 + 0,  ks * 32, BK, lane);
      v16bf a1 = load_frag_a(&As[cur][0][0], wm * 32 + 16, ks * 32, BK, lane);
      v16bf b0 = load_frag_b(&Bs[cur][0][0], wn * 32 + 0,  ks * 32, BK, lane);
      v16bf b1 = load_frag_b(&Bs[cur][0][0], wn * 32 + 16, ks * 32, BK, lane);
      acc[0][0] = WMMA_BF16(a0, b0, acc[0][0]);
      acc[0][1] = WMMA_BF16(a0, b1, acc[0][1]);
      acc[1][0] = WMMA_BF16(a1, b0, acc[1][0]);
      acc[1][1] = WMMA_BF16(a1, b1, acc[1][1]);
    }
    if (wave == 0 && kt + 1 < nk) __builtin_amdgcn_s_wait_tensorcnt(0);
    __syncthreads();
  }

  const int n  = lane & 15;
  const int hi = lane >> 4;
#pragma unroll
  for (int i = 0; i < 2; ++i)
#pragma unroll
    for (int j = 0; j < 2; ++j) {
      const int rbase = tileM + wm * 32 + i * 16 + hi * 8;
      const int ccol  = tileN + wn * 32 + j * 16 + n;
      const float bv  = bias[ccol];
#pragma unroll
      for (int r = 0; r < 8; ++r) {
        const float val = acc[i][j][r] + bv;
        if (OUT_BF16)
          ((__bf16*)Cout)[(size_t)(rbase + r) * N + ccol] = (__bf16)val;
        else
          ((float*)Cout)[(size_t)(rbase + r) * N + ccol] = val;
      }
    }
}

// ---------------------------------------------------------------------------
// Flash attention over bf16 qkv [B*S, 3*D]; head h at column h*64 (+0/1024/2048)
// Block = (128 q-rows, b*h); 8 waves x 16 rows. Q/K tiles by TDM, V staged
// transposed manually (overlaps the K-tile DMA).
// ---------------------------------------------------------------------------
#define QTILE 128
#define KTILE 128

__global__ __launch_bounds__(256)
void mha_flash_attn_wmma(const __bf16* __restrict__ qkv, __bf16* __restrict__ out) {
  __shared__ __bf16 Qs[QTILE][HEAD_DIM];       // 16 KB, TDM
  __shared__ __bf16 Ks[KTILE][HEAD_DIM];       // 16 KB, TDM (B^T layout for QK^T)
  __shared__ __bf16 Vt[HEAD_DIM][KTILE];       // 16 KB, manual transpose
  __shared__ __bf16 Ps[8][16][KTILE];          // 32 KB per-wave P staging

  const int tid  = threadIdx.x;
  const int lane = tid & 31;
  const int wave = tid >> 5;
  const int n    = lane & 15;
  const int hi   = lane >> 4;
  const int b    = blockIdx.y >> 4;
  const int h    = blockIdx.y & 15;
  const int q0   = blockIdx.x * QTILE;

  const size_t rs = 3 * EMBED_DIM;             // 3072-element row stride
  const __bf16* qbase = qkv + (size_t)b * SEQ * rs + h * HEAD_DIM;
  const __bf16* kbase = qbase + EMBED_DIM;
  const __bf16* vbase = qbase + 2 * EMBED_DIM;

  if (wave == 0)                               // Q tile: in flight during setup
    tdm_load_2d_bf16(lds_offset_of(&Qs[0][0]),
                     qbase + (size_t)q0 * rs, HEAD_DIM, QTILE, rs);

  v8f accO[4];
#pragma unroll
  for (int d = 0; d < 4; ++d) accO[d] = (v8f){0.f,0.f,0.f,0.f,0.f,0.f,0.f,0.f};
  float mrow[8], lrow[8];
#pragma unroll
  for (int r = 0; r < 8; ++r) { mrow[r] = -3.0e38f; lrow[r] = 0.f; }

  const float scale = 0.125f;                  // 1/sqrt(64)

  for (int j0 = 0; j0 < SEQ; j0 += KTILE) {
    __syncthreads();                           // previous tile fully consumed
    if (wave == 0)                             // K tile by TDM...
      tdm_load_2d_bf16(lds_offset_of(&Ks[0][0]),
                       kbase + (size_t)j0 * rs, HEAD_DIM, KTILE, rs);
    // ...while all waves stage V transposed (bf16, uint4 = 8 elements)
#pragma unroll
    for (int i = 0; i < 4; ++i) {
      int t4 = tid + i * 256;                  // 0..1023
      int r = t4 >> 3, c8 = (t4 & 7) * 8;
      union { uint4 u; __bf16 h[8]; } vv;
      vv.u = *(const uint4*)(vbase + (size_t)(j0 + r) * rs + c8);
#pragma unroll
      for (int q = 0; q < 8; ++q) Vt[c8 + q][r] = vv.h[q];
    }
    if (wave == 0) __builtin_amdgcn_s_wait_tensorcnt(0);  // Q(+K) landed
    __syncthreads();

    // S = Q K^T : 16 rows x 128 keys per wave
    v8f sc[8];
#pragma unroll
    for (int t = 0; t < 8; ++t) sc[t] = (v8f){0.f,0.f,0.f,0.f,0.f,0.f,0.f,0.f};
#pragma unroll
    for (int ks = 0; ks < 2; ++ks) {
      v16bf aq = load_frag_a(&Qs[0][0], wave * 16, ks * 32, HEAD_DIM, lane);
#pragma unroll
      for (int t = 0; t < 8; ++t) {
        v16bf bk = load_frag_b(&Ks[0][0], t * 16, ks * 32, HEAD_DIM, lane);
        sc[t] = WMMA_BF16(aq, bk, sc[t]);
      }
    }

    // Online softmax (rows r and r+8 live in the two lane halves)
#pragma unroll
    for (int r = 0; r < 8; ++r) {
      float mx = -3.0e38f;
#pragma unroll
      for (int t = 0; t < 8; ++t) mx = fmaxf(mx, sc[t][r] * scale);
#pragma unroll
      for (int msk = 1; msk < 16; msk <<= 1) mx = fmaxf(mx, __shfl_xor(mx, msk, 32));
      const float mnew = fmaxf(mrow[r], mx);
      const float corr = __expf(mrow[r] - mnew);
      float lsum = 0.f;
#pragma unroll
      for (int t = 0; t < 8; ++t) {
        float p = __expf(sc[t][r] * scale - mnew);
        sc[t][r] = p;
        lsum += p;
      }
#pragma unroll
      for (int msk = 1; msk < 16; msk <<= 1) lsum += __shfl_xor(lsum, msk, 32);
      lrow[r] = lrow[r] * corr + lsum;
      mrow[r] = mnew;
#pragma unroll
      for (int d = 0; d < 4; ++d) accO[d][r] *= corr;
    }

    // C-layout -> A-layout via per-wave LDS staging (same-wave DS in-order)
#pragma unroll
    for (int t = 0; t < 8; ++t)
#pragma unroll
      for (int r = 0; r < 8; ++r)
        Ps[wave][r + hi * 8][t * 16 + n] = (__bf16)sc[t][r];

    // O += P @ V
#pragma unroll
    for (int ks = 0; ks < 4; ++ks) {
      v16bf ap = load_frag_a(&Ps[wave][0][0], 0, ks * 32, KTILE, lane);
#pragma unroll
      for (int d = 0; d < 4; ++d) {
        v16bf bv = load_frag_b(&Vt[0][0], d * 16, ks * 32, KTILE, lane);
        accO[d] = WMMA_BF16(ap, bv, accO[d]);
      }
    }
  }

  // Normalize, emit bf16 [B*S, D] for the output projection
#pragma unroll
  for (int d = 0; d < 4; ++d)
#pragma unroll
    for (int r = 0; r < 8; ++r) {
      const int row = q0 + wave * 16 + hi * 8 + r;
      const int col = h * HEAD_DIM + d * 16 + n;
      out[(size_t)(b * SEQ + row) * EMBED_DIM + col] = (__bf16)(accO[d][r] / lrow[r]);
    }
}

// ---------------------------------------------------------------------------
extern "C" void kernel_launch(void* const* d_in, const int* in_sizes, int n_in,
                              void* d_out, int out_size, void* d_ws, size_t ws_size,
                              hipStream_t stream) {
  const float* query = (const float*)d_in[0];
  // d_in[1] (key) and d_in[2] (value) are ignored by the reference.
  const float* W_qkv = (const float*)d_in[3];
  const float* b_qkv = (const float*)d_in[4];
  const float* W_out = (const float*)d_in[5];
  const float* b_out = (const float*)d_in[6];
  float* out = (float*)d_out;

  const int M = BATCH * SEQ;                        // 4096
  const int D = EMBED_DIM;

  // Workspace (all bf16): 48 MB total
  __bf16* qbf   = (__bf16*)d_ws;                    // [4096,1024]
  __bf16* wqkvT = qbf   + (size_t)M * D;            // [3072,1024]
  __bf16* woutT = wqkvT + (size_t)3 * D * D;        // [1024,1024]
  __bf16* qkv   = woutT + (size_t)D * D;            // [4096,3072]
  __bf16* attnb = qkv   + (size_t)M * 3 * D;        // [4096,1024]

  // Prep: convert activations, convert+transpose weights
  cvt_f32_to_bf16<<<(size_t)M * D / 1024, 256, 0, stream>>>(query, qbf);
  cvt_transpose_bf16<<<dim3(3 * D / 32, D / 32), 256, 0, stream>>>(W_qkv, wqkvT, D, 3 * D);
  cvt_transpose_bf16<<<dim3(D / 32, D / 32), 256, 0, stream>>>(W_out, woutT, D, D);

  // QKV projection (bf16 out)
  dim3 g1(3 * D / BN, M / BM);                      // (48, 32)
  mha_gemm_tdm_wmma<true><<<g1, GEMM_THREADS, 0, stream>>>(
      qbf, wqkvT, b_qkv, (void*)qkv, M, 3 * D, D);

  // Flash attention
  dim3 g2(SEQ / QTILE, BATCH * NUM_HEADS);          // (16, 32)
  mha_flash_attn_wmma<<<g2, 256, 0, stream>>>(qkv, attnb);

  // Output projection (f32 out)
  dim3 g3(D / BN, M / BM);                          // (16, 32)
  mha_gemm_tdm_wmma<false><<<g3, GEMM_THREADS, 0, stream>>>(
      attnb, woutT, b_out, (void*)out, M, D, D);
}
